// Model_6528350290017
// MI455X (gfx1250) — matrix operand order repaired and checked by reading, not compiled, not run on hardware
//
#include <hip/hip_runtime.h>
#include <math.h>

// ---------------------------------------------------------------------------
// Problem constants (from the reference)
// ---------------------------------------------------------------------------
#define NRN 20000
#define NPN 4000
#define CDIM 256
#define NEDGE 300000
#define NLBL 150000

// ---------------------------------------------------------------------------
// Types for CDNA5 WMMA
// ---------------------------------------------------------------------------
typedef __attribute__((ext_vector_type(16))) __bf16 bf16x16;
typedef __attribute__((ext_vector_type(8)))  float  f32x8;

__device__ __forceinline__ unsigned short f32_to_bf16(float f) {
  unsigned int u = __float_as_uint(f);
  u += 0x7fffu + ((u >> 16) & 1u);   // round-to-nearest-even
  return (unsigned short)(u >> 16);
}
__device__ __forceinline__ unsigned int pack_bf16(float a, float b) {
  return (unsigned int)f32_to_bf16(a) | ((unsigned int)f32_to_bf16(b) << 16);
}

// ---------------------------------------------------------------------------
// Tiled WMMA GEMM:  Out[M,N] = act( A[M,K] @ B[K,N] + bias + Cadd )
//   A row-major (lda), B row-major KxN (ldb).  f32 -> bf16 LDS tiles,
//   f32 accumulation with v_wmma_f32_16x16x32_bf16.
//   Workgroup tile 128x128, 8 waves (4x2), each wave 32x64 = 2x4 fragments
//   (8 WMMA per 32-K step).  Fast unguarded load path for interior tiles.
// ---------------------------------------------------------------------------
#define TM 128
#define TN 128
#define TK 32
#define LDS_A 34   // padded stride (elements): 17-bank row skew
#define LDS_B 34   // B stored transposed [TN][TK] with padded stride

__launch_bounds__(256, 1)
__global__ void wmma_gemm_kernel(const float* __restrict__ A, int lda,
                                 const float* __restrict__ B, int ldb,
                                 const float* __restrict__ bias,
                                 const float* __restrict__ Cadd, int ldcadd,
                                 float* __restrict__ Out, int ldo,
                                 int M, int N, int K, int act)
{
  __shared__ unsigned short As[TM * LDS_A];   // [128][34] row-major (M,K)
  __shared__ unsigned short Bt[TN * LDS_B];   // [128][34] (N-major, K minor)

  const int tid  = threadIdx.x;
  const int lane = tid & 31;
  const int wid  = tid >> 5;
  const int wm   = wid & 3;   // 0..3 -> 32-row slice
  const int wn   = wid >> 2;  // 0..1 -> 64-col slice
  const int m0   = blockIdx.y * TM;
  const int n0   = blockIdx.x * TN;

  const bool m_full = (m0 + TM <= M);
  const bool n_full = (n0 + TN <= N);

  f32x8 acc[2][4];
#pragma unroll
  for (int mi = 0; mi < 2; ++mi)
#pragma unroll
    for (int ni = 0; ni < 4; ++ni)
#pragma unroll
      for (int r = 0; r < 8; ++r) acc[mi][ni][r] = 0.0f;

  for (int k0 = 0; k0 < K; k0 += TK) {
    const bool k_full = (k0 + TK <= K);

    // ---- stage A tile (128x32): 2048 bf16 pairs, 8 per thread -------------
    if (m_full && k_full) {
#pragma unroll
      for (int i = 0; i < 8; ++i) {
        int p  = tid + i * 256;
        int r  = p >> 4;                 // row 0..127 (16 consecutive tids/row)
        int cp = p & 15;                 // K pair 0..15 -> coalesced
        const float* ap = A + (size_t)(m0 + r) * lda + k0 + cp * 2;
        *(unsigned int*)&As[r * LDS_A + cp * 2] = pack_bf16(ap[0], ap[1]);
      }
    } else {
#pragma unroll
      for (int i = 0; i < 8; ++i) {
        int p  = tid + i * 256;
        int r  = p >> 4;
        int cp = p & 15;
        int gr = m0 + r;
        int gc = k0 + cp * 2;
        float x0 = 0.f, x1 = 0.f;
        if (gr < M) {
          if (gc     < K) x0 = A[(size_t)gr * lda + gc];
          if (gc + 1 < K) x1 = A[(size_t)gr * lda + gc + 1];
        }
        *(unsigned int*)&As[r * LDS_A + cp * 2] = pack_bf16(x0, x1);
      }
    }
    // ---- stage B tile (32x128) transposed into [n][k] ---------------------
    if (n_full && k_full) {
#pragma unroll
      for (int i = 0; i < 8; ++i) {
        int p  = tid + i * 256;
        int kp = p >> 7;                 // K pair 0..15
        int n  = p & 127;                // n contiguous across lanes
        const float* bp = B + (size_t)(k0 + kp * 2) * ldb + n0 + n;
        *(unsigned int*)&Bt[n * LDS_B + kp * 2] = pack_bf16(bp[0], bp[ldb]);
      }
    } else {
#pragma unroll
      for (int i = 0; i < 8; ++i) {
        int p  = tid + i * 256;
        int kp = p >> 7;
        int n  = p & 127;
        int gk = k0 + kp * 2;
        int gn = n0 + n;
        float x0 = 0.f, x1 = 0.f;
        if (gn < N) {
          if (gk     < K) x0 = B[(size_t)gk * ldb + gn];
          if (gk + 1 < K) x1 = B[(size_t)(gk + 1) * ldb + gn];
        }
        *(unsigned int*)&Bt[n * LDS_B + kp * 2] = pack_bf16(x0, x1);
      }
    }
    // ---- prefetch next A K-tile into cache (global_prefetch_b8) -----------
    if (k0 + TK < K) {
      int r  = tid >> 1;
      int gr = m0 + r; if (gr >= M) gr = M - 1;
      __builtin_prefetch(&A[(size_t)gr * lda + (k0 + TK) + (tid & 1) * 16], 0, 1);
    }
    __syncthreads();

    // ---- assemble fragments, k = 16*(lane/16) + s in both A and B ------------------
    union FR { bf16x16 v; unsigned int u[8]; };
    FR afr[2], bfr[4];
#pragma unroll
    for (int mi = 0; mi < 2; ++mi) {
      int row = wm * 32 + mi * 16 + (lane & 15);
      int kb  = (lane >> 4) * 16;                // lanes 16-31: K offset +16
#pragma unroll
      for (int j = 0; j < 8; ++j) {
        int kk = kb + j * 2;
        afr[mi].u[j] = *(const unsigned int*)&As[row * LDS_A + kk];
      }
    }
#pragma unroll
    for (int ni = 0; ni < 4; ++ni) {
      int col = wn * 64 + ni * 16 + (lane & 15);
      int kb  = (lane >> 4) * 16;                // lanes 16-31: K offset +16
#pragma unroll
      for (int j = 0; j < 8; ++j)
        bfr[ni].u[j] = *(const unsigned int*)&Bt[col * LDS_B + kb + j * 2];
    }
#pragma unroll
    for (int mi = 0; mi < 2; ++mi)
#pragma unroll
      for (int ni = 0; ni < 4; ++ni)
        acc[mi][ni] = __builtin_amdgcn_wmma_f32_16x16x32_bf16(
            false, afr[mi].v, false, bfr[ni].v,
            (short)0, acc[mi][ni], false, false);
    __syncthreads();
  }

  // ---- epilogue: bias / Cadd / ReLU, guarded stores -----------------------
#pragma unroll
  for (int mi = 0; mi < 2; ++mi) {
#pragma unroll
    for (int ni = 0; ni < 4; ++ni) {
      int col = n0 + wn * 64 + ni * 16 + (lane & 15);
      float bv = (bias && col < N) ? bias[col] : 0.0f;
#pragma unroll
      for (int r = 0; r < 8; ++r) {
        int row = m0 + wm * 32 + mi * 16 + r + ((lane >> 4) ? 8 : 0);
        if (row < M && col < N) {
          float v = acc[mi][ni][r] + bv;
          if (Cadd) v += Cadd[(size_t)row * ldcadd + col];
          if (act)  v = fmaxf(v, 0.0f);
          Out[(size_t)row * ldo + col] = v;
        }
      }
    }
  }
}

// ---------------------------------------------------------------------------
// Utility kernels
// ---------------------------------------------------------------------------
__global__ void fill_f32(float* __restrict__ p, float v, long n) {
  long i = (long)blockIdx.x * blockDim.x + threadIdx.x;
  if (i < n) p[i] = v;
}

__global__ void gather_rows_kernel(const float* __restrict__ src, int lds_,
                                   const int* __restrict__ idx,
                                   float* __restrict__ dst, int ldd, int col0,
                                   int ncols, int nrows)
{
  int r = blockIdx.x;
  if (r >= nrows) return;
  int s = idx ? idx[r] : r;
  for (int c = threadIdx.x; c < ncols; c += blockDim.x)
    dst[(size_t)r * ldd + col0 + c] = src[(size_t)s * lds_ + c];
}

// per-row L2 normalization of a 256-wide column block; one wave per row
__global__ void l2norm_block_kernel(float* __restrict__ X, int ld, int col0,
                                    int ncols, int nrows)
{
  int r    = blockIdx.x * (blockDim.x >> 5) + (threadIdx.x >> 5);
  int lane = threadIdx.x & 31;
  if (r >= nrows) return;
  float* p = X + (size_t)r * ld + col0;
  float ss = 0.f;
  for (int c = lane; c < ncols; c += 32) { float v = p[c]; ss += v * v; }
#pragma unroll
  for (int off = 16; off; off >>= 1) ss += __shfl_xor(ss, off, 32);
  float scale = 1.0f / fmaxf(sqrtf(ss), 1e-12f);
  for (int c = lane; c < ncols; c += 32) p[c] *= scale;
}

// conv1d(3->32,k=3,pad=1) + ReLU + maxpool2 : out (NPN,32,500)
__global__ void conv1_kernel(const float* __restrict__ prot_x,
                             const float* __restrict__ w,
                             const float* __restrict__ b,
                             float* __restrict__ out)
{
  int n  = blockIdx.x;
  int oc = blockIdx.y;
  const float* in = prot_x + (size_t)n * 19871 + 16871;   // (3,1000)
  float wv[9];
#pragma unroll
  for (int i = 0; i < 9; ++i) wv[i] = w[oc * 9 + i];
  float bb = b[oc];
  for (int o = threadIdx.x; o < 500; o += blockDim.x) {
    float mx = -__builtin_inff();
#pragma unroll
    for (int s = 0; s < 2; ++s) {
      int pos = 2 * o + s;
      float acc = bb;
#pragma unroll
      for (int ic = 0; ic < 3; ++ic)
#pragma unroll
        for (int t = 0; t < 3; ++t) {
          int ip = pos - 1 + t;
          float xv = (ip >= 0 && ip < 1000) ? in[ic * 1000 + ip] : 0.f;
          acc = fmaf(wv[ic * 3 + t], xv, acc);
        }
      mx = fmaxf(mx, fmaxf(acc, 0.f));
    }
    out[((size_t)n * 32 + oc) * 500 + o] = mx;
  }
}

// conv1d(32->64,k=3,pad=1) + ReLU + maxpool2 : out (NPN,64,250)
__global__ void conv2_kernel(const float* __restrict__ h1,
                             const float* __restrict__ w,
                             const float* __restrict__ b,
                             float* __restrict__ out)
{
  __shared__ float ws[96];
  int n = blockIdx.x, oc = blockIdx.y;
  for (int i = threadIdx.x; i < 96; i += blockDim.x) ws[i] = w[oc * 96 + i];
  __syncthreads();
  const float* in = h1 + (size_t)n * 32 * 500;
  float bb = b[oc];
  for (int o = threadIdx.x; o < 250; o += blockDim.x) {
    float mx = -__builtin_inff();
#pragma unroll
    for (int s = 0; s < 2; ++s) {
      int pos = 2 * o + s;
      float acc = bb;
      for (int ic = 0; ic < 32; ++ic) {
        const float* row = in + ic * 500;
#pragma unroll
        for (int t = 0; t < 3; ++t) {
          int ip = pos - 1 + t;
          float xv = (ip >= 0 && ip < 500) ? row[ip] : 0.f;
          acc = fmaf(ws[ic * 3 + t], xv, acc);
        }
      }
      mx = fmaxf(mx, fmaxf(acc, 0.f));
    }
    out[((size_t)n * 64 + oc) * 250 + o] = mx;
  }
}

// edge scatter for segment mean/var: one block (256 thr) per edge
//  mode 0: msg = relu(h)+1e-7   (gen)
//  mode 1: msg = h              (sage)
//  mode 2: msg = alpha * h      (transformer, alpha = aE/max(denom,1e-16))
__global__ void scatter_msg_kernel(const float* __restrict__ H,
                                   const int* __restrict__ sidx,
                                   const int* __restrict__ didx,
                                   int nE, int mode,
                                   const float* __restrict__ aE,
                                   const float* __restrict__ denom,
                                   float* __restrict__ sum,
                                   float* __restrict__ sumsq,
                                   float* __restrict__ cnt)
{
  int e = blockIdx.x;
  if (e >= nE) return;
  int s = sidx[e], d = didx[e];
  int f = threadIdx.x;
  float v = H[(size_t)s * CDIM + f];
  if (mode == 0)      v = fmaxf(v, 0.f) + 1e-7f;
  else if (mode == 2) v = (aE[e] / fmaxf(denom[d], 1e-16f)) * v;
  unsafeAtomicAdd(&sum  [(size_t)d * CDIM + f], v);
  unsafeAtomicAdd(&sumsq[(size_t)d * CDIM + f], v * v);
  if (f == 0) unsafeAtomicAdd(&cnt[d], 1.0f);
}

// out = sqrt(max(E[x^2]-E[x]^2,0)+1e-5) (+T)
__global__ void finish_std_kernel(const float* __restrict__ sum,
                                  const float* __restrict__ sumsq,
                                  const float* __restrict__ cnt,
                                  const float* __restrict__ T, int ldt,
                                  float* __restrict__ out, int ldo, int n)
{
  int i = blockIdx.x;
  if (i >= n) return;
  float c = fmaxf(cnt[i], 1.0f);
  int f = threadIdx.x;
  float m1  = sum  [(size_t)i * CDIM + f] / c;
  float m2  = sumsq[(size_t)i * CDIM + f] / c;
  float var = fmaxf(m2 - m1 * m1, 0.f);
  float v   = sqrtf(var + 1e-5f);
  if (T) v += T[(size_t)i * ldt + f];
  out[(size_t)i * ldo + f] = v;
}

// attention logits: one wave per edge; logit = dot(q[d],k[s])/16; seg-max
__global__ void tr_logits_kernel(const float* __restrict__ q,
                                 const float* __restrict__ k,
                                 const int* __restrict__ sidx,
                                 const int* __restrict__ didx, int nE,
                                 float* __restrict__ logits,
                                 float* __restrict__ mmax)
{
  int e    = blockIdx.x * (blockDim.x >> 5) + (threadIdx.x >> 5);
  int lane = threadIdx.x & 31;
  if (e >= nE) return;
  int s = sidx[e], d = didx[e];
  const float* qp = q + (size_t)d * CDIM;
  const float* kp = k + (size_t)s * CDIM;
  float acc = 0.f;
  for (int c = lane; c < CDIM; c += 32) acc = fmaf(qp[c], kp[c], acc);
#pragma unroll
  for (int off = 16; off; off >>= 1) acc += __shfl_xor(acc, off, 32);
  acc *= 0.0625f;   // 1/sqrt(256)
  if (lane == 0) {
    logits[e] = acc;
    if (acc >= 0.f) atomicMax((int*)&mmax[d], __float_as_int(acc));
    else            atomicMin((unsigned int*)&mmax[d], __float_as_uint(acc));
  }
}

__global__ void tr_exp_kernel(const float* __restrict__ logits,
                              const int* __restrict__ didx,
                              const float* __restrict__ mmax,
                              float* __restrict__ aE,
                              float* __restrict__ denom, int nE)
{
  int e = blockIdx.x * blockDim.x + threadIdx.x;
  if (e >= nE) return;
  int d = didx[e];
  float a = expf(logits[e] - mmax[d]);
  aE[e] = a;
  unsafeAtomicAdd(&denom[d], a);
}

// out[r] = dot(hid[r,0:256], w2) + b2 ; one wave per row
__global__ void final_dot_kernel(const float* __restrict__ hid,
                                 const float* __restrict__ w2,
                                 const float* __restrict__ b2,
                                 float* __restrict__ out, int n)
{
  int r    = blockIdx.x * (blockDim.x >> 5) + (threadIdx.x >> 5);
  int lane = threadIdx.x & 31;
  if (r >= n) return;
  const float* hp = hid + (size_t)r * CDIM;
  float acc = 0.f;
  for (int c = lane; c < CDIM; c += 32) acc = fmaf(hp[c], w2[c], acc);
#pragma unroll
  for (int off = 16; off; off >>= 1) acc += __shfl_xor(acc, off, 32);
  if (lane == 0) out[r] = acc + b2[0];
}

// ---------------------------------------------------------------------------
// Host-side helpers
// ---------------------------------------------------------------------------
static void gemm(hipStream_t s, const float* A, int lda, const float* B, int ldb,
                 const float* bias, const float* Cadd, int ldcadd,
                 float* Out, int ldo, int M, int N, int K, int act)
{
  dim3 grid((unsigned)((N + TN - 1) / TN), (unsigned)((M + TM - 1) / TM));
  wmma_gemm_kernel<<<grid, dim3(256), 0, s>>>(A, lda, B, ldb, bias, Cadd,
                                              ldcadd, Out, ldo, M, N, K, act);
}
static void fillf(hipStream_t s, float* p, float v, long n) {
  long blocks = (n + 255) / 256;
  fill_f32<<<dim3((unsigned)blocks), dim3(256), 0, s>>>(p, v, n);
}
static void l2norm(hipStream_t s, float* X, int ld, int col0, int rows) {
  l2norm_block_kernel<<<dim3((unsigned)((rows + 7) / 8)), dim3(256), 0, s>>>(
      X, ld, col0, CDIM, rows);
}
static void gatherc(hipStream_t s, const float* src, int lds_, const int* idx,
                    float* dst, int ldd, int col0, int rows) {
  gather_rows_kernel<<<dim3((unsigned)rows), dim3(256), 0, s>>>(
      src, lds_, idx, dst, ldd, col0, CDIM, rows);
}

// ---------------------------------------------------------------------------
extern "C" void kernel_launch(void* const* d_in, const int* in_sizes, int n_in,
                              void* d_out, int out_size, void* d_ws, size_t ws_size,
                              hipStream_t stream)
{
  (void)in_sizes; (void)n_in; (void)out_size; (void)ws_size;
  const float* rna_x    = (const float*)d_in[0];
  const float* prot_x   = (const float*)d_in[1];
  const float* emb_rna  = (const float*)d_in[2];
  const float* emb_prot = (const float*)d_in[3];
  const float* W_rna4   = (const float*)d_in[4];
  const float* b_rna4   = (const float*)d_in[5];
  const float* W_rna5   = (const float*)d_in[6];
  const float* b_rna5   = (const float*)d_in[7];
  const float* W_rnam   = (const float*)d_in[8];
  const float* b_rnam   = (const float*)d_in[9];
  const float* W_p5     = (const float*)d_in[10];
  const float* b_p5     = (const float*)d_in[11];
  const float* W_pdist  = (const float*)d_in[12];
  const float* b_pdist  = (const float*)d_in[13];
  const float* conv1_w  = (const float*)d_in[14];
  const float* conv1_b  = (const float*)d_in[15];
  const float* conv2_w  = (const float*)d_in[16];
  const float* conv2_b  = (const float*)d_in[17];
  const float* fc1_w    = (const float*)d_in[18];
  const float* fc1_b    = (const float*)d_in[19];
  const float* gen_Wsrc = (const float*)d_in[20];
  const float* gen_bsrc = (const float*)d_in[21];
  const float* gen_Wdst = (const float*)d_in[22];
  const float* gen_bdst = (const float*)d_in[23];
  const float* gen_W1   = (const float*)d_in[24];
  const float* gen_b1   = (const float*)d_in[25];
  const float* gen_W2   = (const float*)d_in[26];
  const float* gen_b2   = (const float*)d_in[27];
  const float* sage_Wl  = (const float*)d_in[28];
  const float* sage_bl  = (const float*)d_in[29];
  const float* sage_Wr  = (const float*)d_in[30];
  const float* tr_Wq    = (const float*)d_in[31];
  const float* tr_bq    = (const float*)d_in[32];
  const float* tr_Wk    = (const float*)d_in[33];
  const float* tr_bk    = (const float*)d_in[34];
  const float* tr_Wv    = (const float*)d_in[35];
  const float* tr_bv    = (const float*)d_in[36];
  const float* tr_Ws    = (const float*)d_in[37];
  const float* tr_bs    = (const float*)d_in[38];
  const float* cls_W1   = (const float*)d_in[39];
  const float* cls_b1   = (const float*)d_in[40];
  const float* cls_W2   = (const float*)d_in[41];
  const float* cls_b2   = (const float*)d_in[42];
  const int*   rna_id   = (const int*)d_in[43];
  const int*   prot_id  = (const int*)d_in[44];
  const int*   e_src    = (const int*)d_in[45];            // edge_index[0]
  const int*   e_dst    = ((const int*)d_in[45]) + NEDGE;  // edge_index[1]
  const int*   l_src    = (const int*)d_in[46];            // label rna idx
  const int*   l_dst    = ((const int*)d_in[46]) + NLBL;   // label prot idx
  float* out = (float*)d_out;

  // ---- workspace bump allocator ----
  char*  base = (char*)d_ws;
  size_t cur  = 0;
  auto alloc = [&](size_t elems) -> float* {
    float* p = (float*)(base + cur);
    cur += ((elems * sizeof(float) + 255) & ~(size_t)255);
    return p;
  };
  float* x_r   = alloc((size_t)NRN * 1024);
  float* x_p   = alloc((size_t)NPN * 1024);
  float* xr    = alloc((size_t)NRN * CDIM);
  float* xp    = alloc((size_t)NPN * CDIM);
  float* xr2   = alloc((size_t)NRN * CDIM);
  float* xp2   = alloc((size_t)NPN * CDIM);
  float* xr3   = alloc((size_t)NRN * CDIM);
  float* xp3   = alloc((size_t)NPN * CDIM);
  float* tA    = alloc((size_t)NRN * 512);   // wide temp (gen MLP mid / T)
  float* tB    = alloc((size_t)NRN * CDIM);
  float* tC    = alloc((size_t)NRN * CDIM);
  float* tD    = alloc((size_t)NRN * CDIM);
  float* sum_r = alloc((size_t)NRN * CDIM);
  float* sq_r  = alloc((size_t)NRN * CDIM);
  float* cnt_r = alloc((size_t)NRN);
  float* sum_p = alloc((size_t)NPN * CDIM);
  float* sq_p  = alloc((size_t)NPN * CDIM);
  float* cnt_p = alloc((size_t)NPN);
  float* logit = alloc((size_t)NEDGE);
  float* aE    = alloc((size_t)NEDGE);
  float* den_r = alloc((size_t)NRN);
  float* den_p = alloc((size_t)NPN);
  float* mmx_r = alloc((size_t)NRN);
  float* mmx_p = alloc((size_t)NPN);
  float* c1buf = alloc((size_t)NPN * 32 * 500);
  float* c2buf = alloc((size_t)NPN * 64 * 250);
  float* feat  = alloc((size_t)NLBL * 512);
  float* hid   = alloc((size_t)NLBL * CDIM);

  // ================= node feature construction =================
  // x_r = [l2(r4@W4+b), l2(r5@W5+b), l2(rm@Wm+b), l2(emb_rna)]
  gemm(stream, rna_x + 0,    2474, W_rna4, CDIM, b_rna4, nullptr, 0,
       x_r + 0,   1024, NRN, CDIM, 256, 0);
  l2norm(stream, x_r, 1024, 0, NRN);
  gemm(stream, rna_x + 256,  2474, W_rna5, CDIM, b_rna5, nullptr, 0,
       x_r + 256, 1024, NRN, CDIM, 1024, 0);
  l2norm(stream, x_r, 1024, 256, NRN);
  gemm(stream, rna_x + 1280, 2474, W_rnam, CDIM, b_rnam, nullptr, 0,
       x_r + 512, 1024, NRN, CDIM, 1194, 0);
  l2norm(stream, x_r, 1024, 512, NRN);
  gatherc(stream, emb_rna, CDIM, rna_id, x_r, 1024, 768, NRN);
  l2norm(stream, x_r, 1024, 768, NRN);

  // x_p chunk 0: p5 @ W_p5   (K = 16807)
  gemm(stream, prot_x + 0, 19871, W_p5, CDIM, b_p5, nullptr, 0,
       x_p + 0, 1024, NPN, CDIM, 16807, 0);
  l2norm(stream, x_p, 1024, 0, NPN);
  // x_p chunk 1: conv tower + fc1
  conv1_kernel<<<dim3(NPN, 32), dim3(256), 0, stream>>>(prot_x, conv1_w,
                                                        conv1_b, c1buf);
  conv2_kernel<<<dim3(NPN, 64), dim3(256), 0, stream>>>(c1buf, conv2_w,
                                                        conv2_b, c2buf);
  gemm(stream, c2buf, 16000, fc1_w, CDIM, fc1_b, nullptr, 0,
       x_p + 256, 1024, NPN, CDIM, 16000, 0);
  l2norm(stream, x_p, 1024, 256, NPN);
  // x_p chunk 2: pdist
  gemm(stream, prot_x + 16807, 19871, W_pdist, CDIM, b_pdist, nullptr, 0,
       x_p + 512, 1024, NPN, CDIM, 64, 0);
  l2norm(stream, x_p, 1024, 512, NPN);
  // x_p chunk 3: emb_prot
  gatherc(stream, emb_prot, CDIM, prot_id, x_p, 1024, 768, NPN);
  l2norm(stream, x_p, 1024, 768, NPN);

  // ================= GENConv layer 0 (rna -> prot), out xp =================
  gemm(stream, x_r, 1024, gen_Wsrc, CDIM, gen_bsrc, nullptr, 0,
       tB, CDIM, NRN, CDIM, 1024, 0);                       // h_src
  fillf(stream, sum_p, 0.f, (long)NPN * CDIM);
  fillf(stream, sq_p,  0.f, (long)NPN * CDIM);
  fillf(stream, cnt_p, 0.f, NPN);
  scatter_msg_kernel<<<dim3(NEDGE), dim3(CDIM), 0, stream>>>(
      tB, e_src, e_dst, NEDGE, 0, nullptr, nullptr, sum_p, sq_p, cnt_p);
  gemm(stream, x_p, 1024, gen_Wdst, CDIM, gen_bdst, nullptr, 0,
       tC, CDIM, NPN, CDIM, 1024, 0);                       // dst term
  finish_std_kernel<<<dim3(NPN), dim3(CDIM), 0, stream>>>(
      sum_p, sq_p, cnt_p, tC, CDIM, tD, CDIM, NPN);
  gemm(stream, tD, CDIM, gen_W1, 512, gen_b1, nullptr, 0,
       tA, 512, NPN, 512, CDIM, 1);                         // relu MLP mid
  gemm(stream, tA, 512, gen_W2, CDIM, gen_b2, nullptr, 0,
       xp, CDIM, NPN, CDIM, 512, 1);                        // relu(_gen)

  // ================= GENConv layer 1 (prot -> rna), out xr =================
  gemm(stream, x_p, 1024, gen_Wsrc + 1024 * CDIM, CDIM, gen_bsrc + CDIM,
       nullptr, 0, tB, CDIM, NPN, CDIM, 1024, 0);
  fillf(stream, sum_r, 0.f, (long)NRN * CDIM);
  fillf(stream, sq_r,  0.f, (long)NRN * CDIM);
  fillf(stream, cnt_r, 0.f, NRN);
  scatter_msg_kernel<<<dim3(NEDGE), dim3(CDIM), 0, stream>>>(
      tB, e_dst, e_src, NEDGE, 0, nullptr, nullptr, sum_r, sq_r, cnt_r);
  gemm(stream, x_r, 1024, gen_Wdst + 1024 * CDIM, CDIM, gen_bdst + CDIM,
       nullptr, 0, tC, CDIM, NRN, CDIM, 1024, 0);
  finish_std_kernel<<<dim3(NRN), dim3(CDIM), 0, stream>>>(
      sum_r, sq_r, cnt_r, tC, CDIM, tD, CDIM, NRN);
  gemm(stream, tD, CDIM, gen_W1 + CDIM * 512, 512, gen_b1 + 512, nullptr, 0,
       tA, 512, NRN, 512, CDIM, 1);
  gemm(stream, tA, 512, gen_W2 + 512 * CDIM, CDIM, gen_b2 + CDIM, nullptr, 0,
       xr, CDIM, NRN, CDIM, 512, 1);

  // ================= SAGE layer 0 (xr -> xp2) =================
  fillf(stream, sum_p, 0.f, (long)NPN * CDIM);
  fillf(stream, sq_p,  0.f, (long)NPN * CDIM);
  fillf(stream, cnt_p, 0.f, NPN);
  scatter_msg_kernel<<<dim3(NEDGE), dim3(CDIM), 0, stream>>>(
      xr, e_src, e_dst, NEDGE, 1, nullptr, nullptr, sum_p, sq_p, cnt_p);
  finish_std_kernel<<<dim3(NPN), dim3(CDIM), 0, stream>>>(
      sum_p, sq_p, cnt_p, nullptr, 0, tD, CDIM, NPN);       // aggr
  gemm(stream, xp, CDIM, sage_Wr, CDIM, nullptr, nullptr, 0,
       tC, CDIM, NPN, CDIM, CDIM, 0);                       // x@Wr
  gemm(stream, tD, CDIM, sage_Wl, CDIM, sage_bl, tC, CDIM,
       xp2, CDIM, NPN, CDIM, CDIM, 1);                      // relu(sum)

  // ================= SAGE layer 1 (xp -> xr2) =================
  fillf(stream, sum_r, 0.f, (long)NRN * CDIM);
  fillf(stream, sq_r,  0.f, (long)NRN * CDIM);
  fillf(stream, cnt_r, 0.f, NRN);
  scatter_msg_kernel<<<dim3(NEDGE), dim3(CDIM), 0, stream>>>(
      xp, e_dst, e_src, NEDGE, 1, nullptr, nullptr, sum_r, sq_r, cnt_r);
  finish_std_kernel<<<dim3(NRN), dim3(CDIM), 0, stream>>>(
      sum_r, sq_r, cnt_r, nullptr, 0, tD, CDIM, NRN);
  gemm(stream, xr, CDIM, sage_Wr + CDIM * CDIM, CDIM, nullptr, nullptr, 0,
       tC, CDIM, NRN, CDIM, CDIM, 0);
  gemm(stream, tD, CDIM, sage_Wl + CDIM * CDIM, CDIM, sage_bl + CDIM, tC, CDIM,
       xr2, CDIM, NRN, CDIM, CDIM, 1);

  // ================= TransformerConv 0 (xr2 -> xp3) =================
  gemm(stream, xp2, CDIM, tr_Wq, CDIM, tr_bq, nullptr, 0, tC, CDIM, NPN, CDIM, CDIM, 0); // q (dst)
  gemm(stream, xr2, CDIM, tr_Wk, CDIM, tr_bk, nullptr, 0, tB, CDIM, NRN, CDIM, CDIM, 0); // k (src)
  gemm(stream, xr2, CDIM, tr_Wv, CDIM, tr_bv, nullptr, 0, tD, CDIM, NRN, CDIM, CDIM, 0); // v (src)
  fillf(stream, mmx_p, -__builtin_inff(), NPN);
  fillf(stream, den_p, 0.f, NPN);
  fillf(stream, sum_p, 0.f, (long)NPN * CDIM);
  fillf(stream, sq_p,  0.f, (long)NPN * CDIM);
  fillf(stream, cnt_p, 0.f, NPN);
  tr_logits_kernel<<<dim3((NEDGE + 7) / 8), dim3(256), 0, stream>>>(
      tC, tB, e_src, e_dst, NEDGE, logit, mmx_p);
  tr_exp_kernel<<<dim3((NEDGE + 255) / 256), dim3(256), 0, stream>>>(
      logit, e_dst, mmx_p, aE, den_p, NEDGE);
  scatter_msg_kernel<<<dim3(NEDGE), dim3(CDIM), 0, stream>>>(
      tD, e_src, e_dst, NEDGE, 2, aE, den_p, sum_p, sq_p, cnt_p);
  gemm(stream, xp2, CDIM, tr_Ws, CDIM, tr_bs, nullptr, 0,
       tA, CDIM, NPN, CDIM, CDIM, 0);                       // skip term
  finish_std_kernel<<<dim3(NPN), dim3(CDIM), 0, stream>>>(
      sum_p, sq_p, cnt_p, tA, CDIM, xp3, CDIM, NPN);

  // ================= TransformerConv 1 (xp2 -> xr3) =================
  gemm(stream, xr2, CDIM, tr_Wq + CDIM * CDIM, CDIM, tr_bq + CDIM, nullptr, 0,
       tC, CDIM, NRN, CDIM, CDIM, 0);                       // q (dst = rna)
  gemm(stream, xp2, CDIM, tr_Wk + CDIM * CDIM, CDIM, tr_bk + CDIM, nullptr, 0,
       tB, CDIM, NPN, CDIM, CDIM, 0);                       // k (src = prot)
  gemm(stream, xp2, CDIM, tr_Wv + CDIM * CDIM, CDIM, tr_bv + CDIM, nullptr, 0,
       tD, CDIM, NPN, CDIM, CDIM, 0);                       // v (src = prot)
  fillf(stream, mmx_r, -__builtin_inff(), NRN);
  fillf(stream, den_r, 0.f, NRN);
  fillf(stream, sum_r, 0.f, (long)NRN * CDIM);
  fillf(stream, sq_r,  0.f, (long)NRN * CDIM);
  fillf(stream, cnt_r, 0.f, NRN);
  tr_logits_kernel<<<dim3((NEDGE + 7) / 8), dim3(256), 0, stream>>>(
      tC, tB, e_dst, e_src, NEDGE, logit, mmx_r);
  tr_exp_kernel<<<dim3((NEDGE + 255) / 256), dim3(256), 0, stream>>>(
      logit, e_src, mmx_r, aE, den_r, NEDGE);
  scatter_msg_kernel<<<dim3(NEDGE), dim3(CDIM), 0, stream>>>(
      tD, e_dst, e_src, NEDGE, 2, aE, den_r, sum_r, sq_r, cnt_r);
  gemm(stream, xr2, CDIM, tr_Ws + CDIM * CDIM, CDIM, tr_bs + CDIM, nullptr, 0,
       tA, CDIM, NRN, CDIM, CDIM, 0);
  finish_std_kernel<<<dim3(NRN), dim3(CDIM), 0, stream>>>(
      sum_r, sq_r, cnt_r, tA, CDIM, xr3, CDIM, NRN);

  // ================= classifier =================
  gatherc(stream, xr3, CDIM, l_src, feat, 512, 0,   NLBL);
  gatherc(stream, xp3, CDIM, l_dst, feat, 512, 256, NLBL);
  gemm(stream, feat, 512, cls_W1, CDIM, cls_b1, nullptr, 0,
       hid, CDIM, NLBL, CDIM, 512, 1);                      // relu hidden
  final_dot_kernel<<<dim3((NLBL + 7) / 8), dim3(256), 0, stream>>>(
      hid, cls_W2, cls_b2, out, NLBL);
}
